// Seq2Seq_35742717837713
// MI455X (gfx1250) — compile-verified, matrix operand order checked by reading, not run on hardware
//
#include <hip/hip_runtime.h>
#include <hip/hip_bf16.h>

typedef __attribute__((ext_vector_type(16))) __bf16 v16bf;
typedef __attribute__((ext_vector_type(8)))  __bf16 v8bf;
typedef __attribute__((ext_vector_type(8)))  float  v8f;

#define SEQ_LEN 512
#define BATCH   1024
#define IN_SIZE 3
#define HID     512
#define HID3    1536
#define HORIZON 96

// ---------------------------------------------------------------------------
// Utility kernels
// ---------------------------------------------------------------------------
__global__ void k_f32_to_bf16(const float* __restrict__ in, __bf16* __restrict__ out, int n) {
    int i = blockIdx.x * blockDim.x + threadIdx.x;
    if (i < n) out[i] = (__bf16)in[i];
}

__global__ void k_zero_h(float* __restrict__ hf, __bf16* __restrict__ hb, int n) {
    int i = blockIdx.x * blockDim.x + threadIdx.x;
    if (i < n) { hf[i] = 0.0f; hb[i] = (__bf16)0.0f; }
}

// dec_in0 = src[-1, :, 0]
__global__ void k_init_decin(const float* __restrict__ src, float* __restrict__ dec_in) {
    int m = blockIdx.x * blockDim.x + threadIdx.x;
    if (m < BATCH) dec_in[m] = src[(size_t)(SEQ_LEN - 1) * BATCH * IN_SIZE + m * IN_SIZE + 0];
}

// ---------------------------------------------------------------------------
// WMMA GEMM:  C[M x N] (f32) = A[M x K] (bf16, row-major) * W[N x K]^T (bf16)
// Block = 256 threads = 8 waves.  Each wave computes a 32x32 output tile
// (2 M-tiles x 2 N-tiles = 4 WMMAs per k-step; 8 b128 loads feed 4 WMMAs ->
// 16 FLOP/byte operand reuse).  Block covers 64 rows x 128 cols.
// Grid = (M/64, N/128).
// Fragment layout per ISA 7.12.2 (16-bit A 16x32): lane&15 selects the row
// (A) / column (B=W row); the 16 bf16 elements per lane are two contiguous
// 8-element chunks at k = k0 + 8*(lane>=16) and +16.
// ---------------------------------------------------------------------------
__global__ __launch_bounds__(256)
void k_wmma_gemm(const __bf16* __restrict__ A, const __bf16* __restrict__ W,
                 float* __restrict__ C, int K, int N) {
    const int tid   = threadIdx.x;
    const int lane  = tid & 31;
    const int wave  = tid >> 5;          // 0..7
    const int ln    = lane & 15;
    const int khalf = (lane >> 4) << 3;  // 0 or 8

    const int mbase = blockIdx.x * 64 + (wave >> 2) * 32;   // 2 waves down M
    const int nbase = blockIdx.y * 128 + (wave & 3) * 32;   // 4 waves across N

    const __bf16* Arow0 = A + (size_t)(mbase + ln) * K;
    const __bf16* Arow1 = A + (size_t)(mbase + 16 + ln) * K;
    const __bf16* Wrow0 = W + (size_t)(nbase + ln) * K;
    const __bf16* Wrow1 = W + (size_t)(nbase + 16 + ln) * K;

    v8f acc00 = {};   // (m0, n0)
    v8f acc01 = {};   // (m0, n1)
    v8f acc10 = {};   // (m1, n0)
    v8f acc11 = {};   // (m1, n1)

    #pragma unroll 2
    for (int k0 = 0; k0 < K; k0 += 32) {
        const int kofs = k0 + khalf;

        v8bf a0_lo = *(const v8bf*)(Arow0 + kofs);
        v8bf a0_hi = *(const v8bf*)(Arow0 + kofs + 16);
        v16bf a0 = __builtin_shufflevector(a0_lo, a0_hi,
                       0,1,2,3,4,5,6,7,8,9,10,11,12,13,14,15);

        v8bf a1_lo = *(const v8bf*)(Arow1 + kofs);
        v8bf a1_hi = *(const v8bf*)(Arow1 + kofs + 16);
        v16bf a1 = __builtin_shufflevector(a1_lo, a1_hi,
                       0,1,2,3,4,5,6,7,8,9,10,11,12,13,14,15);

        v8bf b0_lo = *(const v8bf*)(Wrow0 + kofs);
        v8bf b0_hi = *(const v8bf*)(Wrow0 + kofs + 16);
        v16bf b0 = __builtin_shufflevector(b0_lo, b0_hi,
                       0,1,2,3,4,5,6,7,8,9,10,11,12,13,14,15);

        v8bf b1_lo = *(const v8bf*)(Wrow1 + kofs);
        v8bf b1_hi = *(const v8bf*)(Wrow1 + kofs + 16);
        v16bf b1 = __builtin_shufflevector(b1_lo, b1_hi,
                       0,1,2,3,4,5,6,7,8,9,10,11,12,13,14,15);

        acc00 = __builtin_amdgcn_wmma_f32_16x16x32_bf16(
                    false, a0, false, b0, (short)0, acc00, false, false);
        acc01 = __builtin_amdgcn_wmma_f32_16x16x32_bf16(
                    false, a0, false, b1, (short)0, acc01, false, false);
        acc10 = __builtin_amdgcn_wmma_f32_16x16x32_bf16(
                    false, a1, false, b0, (short)0, acc10, false, false);
        acc11 = __builtin_amdgcn_wmma_f32_16x16x32_bf16(
                    false, a1, false, b1, (short)0, acc11, false, false);
    }

    // C/D layout: VGPR r -> row = tile_mbase + r + 8*(lane>=16); col = tile_nbase + (lane&15)
    const int row0 = mbase + ((lane >> 4) << 3);
    const int row1 = row0 + 16;
    const int col  = nbase + ln;
    #pragma unroll
    for (int r = 0; r < 8; ++r) {
        C[(size_t)(row0 + r) * N + col]      = acc00[r];
        C[(size_t)(row0 + r) * N + col + 16] = acc01[r];
        C[(size_t)(row1 + r) * N + col]      = acc10[r];
        C[(size_t)(row1 + r) * N + col + 16] = acc11[r];
    }
}

// ---------------------------------------------------------------------------
// Fused GRU gate update:
//   gi = x @ wih.T + bih   (xdim = 3 encoder / 1 decoder; tiny K -> VALU)
//   gh (precomputed h@whh.T) + bhh
//   r = sig(gi_r+gh_r); z = sig(gi_z+gh_z); n = tanh(gi_n + r*gh_n)
//   h' = (1-z)*n + z*h     -> fp32 state + bf16 mirror for next WMMA GEMM
// One thread per (m, j); grid = BATCH*HID/256.
// ---------------------------------------------------------------------------
__global__ __launch_bounds__(256)
void k_gru_update(const float* __restrict__ gh, const float* __restrict__ x, int xdim,
                  const float* __restrict__ wih, const float* __restrict__ bih,
                  const float* __restrict__ bhh,
                  float* __restrict__ hf, __bf16* __restrict__ hb) {
    const int idx = blockIdx.x * blockDim.x + threadIdx.x;
    const int m = idx >> 9;          // /HID
    const int j = idx & (HID - 1);

    float gr = bih[j];
    float gz = bih[HID + j];
    float gn = bih[2 * HID + j];
    for (int k = 0; k < xdim; ++k) {
        const float xv = x[m * xdim + k];
        gr += xv * wih[j * xdim + k];
        gz += xv * wih[(HID + j) * xdim + k];
        gn += xv * wih[(2 * HID + j) * xdim + k];
    }

    const size_t gbase = (size_t)m * HID3;
    const float hr = gh[gbase + j]            + bhh[j];
    const float hz = gh[gbase + HID + j]      + bhh[HID + j];
    const float hn = gh[gbase + 2 * HID + j]  + bhh[2 * HID + j];

    const float r = 1.0f / (1.0f + __expf(-(gr + hr)));
    const float z = 1.0f / (1.0f + __expf(-(gz + hz)));
    const float n = tanhf(gn + r * hn);

    const float hold = hf[idx];
    const float hnew = (1.0f - z) * n + z * hold;
    hf[idx] = hnew;
    hb[idx] = (__bf16)hnew;
}

// ---------------------------------------------------------------------------
// FC head: y[m] = sum_j relu(pre[m][j] + b1[j]) * w2[j] + b2
// One wave per row, shuffle reduction.  Writes d_out[t] and next dec_in.
// ---------------------------------------------------------------------------
__global__ __launch_bounds__(256)
void k_fc_head(const float* __restrict__ pre, const float* __restrict__ b1,
               const float* __restrict__ w2, const float* __restrict__ b2,
               float* __restrict__ out_t, float* __restrict__ dec_in) {
    const int wave = threadIdx.x >> 5;
    const int lane = threadIdx.x & 31;
    const int m = blockIdx.x * 8 + wave;

    float s = 0.0f;
    const float* prow = pre + (size_t)m * HID;
    for (int j = lane; j < HID; j += 32) {
        float a = prow[j] + b1[j];
        a = a > 0.0f ? a : 0.0f;
        s += a * w2[j];
    }
    #pragma unroll
    for (int off = 16; off > 0; off >>= 1)
        s += __shfl_down(s, off, 32);

    if (lane == 0) {
        const float y = s + b2[0];
        out_t[m]  = y;
        dec_in[m] = y;
    }
}

// ---------------------------------------------------------------------------
// Host orchestration
// ---------------------------------------------------------------------------
extern "C" void kernel_launch(void* const* d_in, const int* in_sizes, int n_in,
                              void* d_out, int out_size, void* d_ws, size_t ws_size,
                              hipStream_t stream) {
    const float* src     = (const float*)d_in[0];
    const float* enc_wih = (const float*)d_in[1];
    const float* enc_whh = (const float*)d_in[2];
    const float* enc_bih = (const float*)d_in[3];
    const float* enc_bhh = (const float*)d_in[4];
    const float* dec_wih = (const float*)d_in[5];
    const float* dec_whh = (const float*)d_in[6];
    const float* dec_bih = (const float*)d_in[7];
    const float* dec_bhh = (const float*)d_in[8];
    const float* w1      = (const float*)d_in[9];
    const float* b1      = (const float*)d_in[10];
    const float* w2      = (const float*)d_in[11];
    const float* b2      = (const float*)d_in[12];
    float*       out     = (float*)d_out;

    // workspace layout (256B aligned regions)
    char* ws = (char*)d_ws;
    size_t off = 0;
    auto take = [&](size_t bytes) { char* p = ws + off; off = (off + bytes + 255) & ~(size_t)255; return p; };
    float*  hf       = (float*) take((size_t)BATCH * HID * 4);   // fp32 hidden state
    __bf16* hb       = (__bf16*)take((size_t)BATCH * HID * 2);   // bf16 mirror (GEMM A)
    float*  gh       = (float*) take((size_t)BATCH * HID3 * 4);  // h@whh.T  (also FC1 pre-act)
    __bf16* encW     = (__bf16*)take((size_t)HID3 * HID * 2);
    __bf16* decW     = (__bf16*)take((size_t)HID3 * HID * 2);
    __bf16* w1b      = (__bf16*)take((size_t)HID * HID * 2);
    float*  dec_in   = (float*) take((size_t)BATCH * 4);

    const int NT = 256;

    // one-time prep (graph-safe, deterministic)
    k_f32_to_bf16<<<(HID3 * HID + NT - 1) / NT, NT, 0, stream>>>(enc_whh, encW, HID3 * HID);
    k_f32_to_bf16<<<(HID3 * HID + NT - 1) / NT, NT, 0, stream>>>(dec_whh, decW, HID3 * HID);
    k_f32_to_bf16<<<(HID * HID  + NT - 1) / NT, NT, 0, stream>>>(w1,      w1b,  HID * HID);
    k_zero_h<<<(BATCH * HID + NT - 1) / NT, NT, 0, stream>>>(hf, hb, BATCH * HID);

    const dim3 gemm_grid_gates(BATCH / 64, HID3 / 128);  // 16 x 12
    const dim3 gemm_grid_fc1  (BATCH / 64, HID  / 128);  // 16 x 4
    const int  gru_blocks = (BATCH * HID) / NT;          // 2048

    // ---- encoder: 512 sequential GRU steps ----
    for (int t = 0; t < SEQ_LEN; ++t) {
        k_wmma_gemm<<<gemm_grid_gates, NT, 0, stream>>>(hb, encW, gh, HID, HID3);
        const float* x_t = src + (size_t)t * BATCH * IN_SIZE;
        k_gru_update<<<gru_blocks, NT, 0, stream>>>(gh, x_t, IN_SIZE,
                                                    enc_wih, enc_bih, enc_bhh, hf, hb);
    }

    // ---- decoder: 96 steps ----
    k_init_decin<<<(BATCH + NT - 1) / NT, NT, 0, stream>>>(src, dec_in);
    for (int t = 0; t < HORIZON; ++t) {
        k_wmma_gemm<<<gemm_grid_gates, NT, 0, stream>>>(hb, decW, gh, HID, HID3);
        k_gru_update<<<gru_blocks, NT, 0, stream>>>(gh, dec_in, 1,
                                                    dec_wih, dec_bih, dec_bhh, hf, hb);
        // FC1 pre-activation into gh region (consumed already this step)
        k_wmma_gemm<<<gemm_grid_fc1, NT, 0, stream>>>(hb, w1b, gh, HID, HID);
        k_fc_head<<<BATCH / 8, NT, 0, stream>>>(gh, b1, w2, b2,
                                                out + (size_t)t * BATCH, dec_in);
    }
}
